// SelectiveSSMLayer_40638980555397
// MI455X (gfx1250) — compile-verified, matrix operand order checked by reading, not run on hardware
//
#include <hip/hip_runtime.h>
#include <hip/hip_bf16.h>
#include <math.h>

// ---------------- problem constants ----------------
#define D_MODEL  1024
#define D_INNER  2048
#define DT_RANK  32
#define BATCH    4
#define SEQ      4096
#define NTOK     (BATCH * SEQ)   // 16384 tokens
#define N_XZ     (2 * D_INNER)   // 4096

// ---------------- CDNA5 WMMA types ----------------
typedef __bf16 bf16;
typedef __attribute__((ext_vector_type(16))) __bf16 v16bf;
typedef __attribute__((ext_vector_type(8)))  __bf16 v8bf;
typedef __attribute__((ext_vector_type(8)))  float  v8f;

#define KBLK     32                 // WMMA K per step (16x16x32 bf16)
#define LDSTR    40                 // padded LDS row stride (elems) -> 80B rows, 16B-aligned chunks
#define BUFELE   (128 * LDSTR)      // elements per LDS tile buffer
#define BUFBYTES (BUFELE * 2)       // bytes per LDS tile buffer (10240)

__device__ __forceinline__ float silu_f(float x) {
    return x / (1.0f + __expf(-x));
}
__device__ __forceinline__ float softplus_f(float x) {
    return (x > 20.0f) ? x : log1pf(__expf(x));
}

// Build a 16x32 bf16 A/B fragment from a row-major row pointer.
// CDNA5 layout: lane holds row (lane&15); kb=(lane>>4)*8;
// elements 0..7 = K kb..kb+7, elements 8..15 = K kb+16..kb+23.
__device__ __forceinline__ v16bf make_frag(const bf16* rowptr, int kb) {
    union { v16bf v; v8bf h[2]; } u;
    u.h[0] = *(const v8bf*)(rowptr + kb);
    u.h[1] = *(const v8bf*)(rowptr + kb + 16);
    return u.v;
}

// Low 32 bits of a flat shared pointer == LDS byte offset (aperture decode).
__device__ __forceinline__ uint32_t lds_off(const void* p) {
    return (uint32_t)(uintptr_t)p;
}

// CDNA5 async DMA: each lane copies 16B global -> LDS, tracked by ASYNCcnt.
__device__ __forceinline__ void async_b128(uint32_t lds_byte, const bf16* g) {
    asm volatile("global_load_async_to_lds_b128 %0, %1, off"
                 :: "v"(lds_byte), "v"(g) : "memory");
}

// ---------------- shared WMMA main loop (double-buffered async pipeline) ----
// Block tile 128x128, 8 waves (2 m x 4 n), each wave 64x32 = 4x2 fragments.
template <int K>
__device__ __forceinline__ void wmma_mainloop(const bf16* __restrict__ A,
                                              const bf16* __restrict__ W,
                                              int m0, int n0,
                                              bf16* As, bf16* Bs,   // [2][BUFELE]
                                              v8f acc[4][2]) {
    const int tid  = threadIdx.x;
    const int lane = tid & 31;
    const int wid  = tid >> 5;
    const int wm   = (wid & 1) * 64;
    const int wn   = (wid >> 1) * 32;
    const int kb   = (lane >> 4) * 8;
    const int lr   = lane & 15;

    // 512 chunks of 8 elems (16B) per 128x32 tile; thread owns chunks tid, tid+256.
    const int ra0 = tid >> 2,         oa0 = (tid & 3) * 8;
    const int ra1 = (tid + 256) >> 2, oa1 = ((tid + 256) & 3) * 8;

    const bf16* gA0 = A + (size_t)(m0 + ra0) * K + oa0;
    const bf16* gA1 = A + (size_t)(m0 + ra1) * K + oa1;
    const bf16* gB0 = W + (size_t)(n0 + ra0) * K + oa0;
    const bf16* gB1 = W + (size_t)(n0 + ra1) * K + oa1;

    const uint32_t lA0 = lds_off(As + ra0 * LDSTR + oa0);
    const uint32_t lA1 = lds_off(As + ra1 * LDSTR + oa1);
    const uint32_t lB0 = lds_off(Bs + ra0 * LDSTR + oa0);
    const uint32_t lB1 = lds_off(Bs + ra1 * LDSTR + oa1);

    constexpr int NSTEP = K / KBLK;

    // prologue: fill buffer 0 (4 async ops per wave)
    async_b128(lA0, gA0);
    async_b128(lA1, gA1);
    async_b128(lB0, gB0);
    async_b128(lB1, gB1);

    for (int i = 0; i < NSTEP; ++i) {
        const int cur = i & 1;
        if (i + 1 < NSTEP) {
            // prefetch next k-block into the other buffer (safe: trailing barrier
            // of the previous iteration guarantees everyone is done reading it)
            const uint32_t nb = (uint32_t)(cur ^ 1) * BUFBYTES;
            const int koff = (i + 1) * KBLK;
            async_b128(lA0 + nb, gA0 + koff);
            async_b128(lA1 + nb, gA1 + koff);
            async_b128(lB0 + nb, gB0 + koff);
            async_b128(lB1 + nb, gB1 + koff);
            // wait for the 4 oldest ops (current buffer); 4 newest stay in flight
            asm volatile("s_wait_asynccnt 0x4" ::: "memory");
        } else {
            asm volatile("s_wait_asynccnt 0x0" ::: "memory");
        }
        __syncthreads();   // all waves' DMA for `cur` landed

        const bf16* Ab = As + cur * BUFELE;
        const bf16* Bb = Bs + cur * BUFELE;
        v16bf bfr[2];
        #pragma unroll
        for (int ni = 0; ni < 2; ++ni)
            bfr[ni] = make_frag(Bb + (wn + ni * 16 + lr) * LDSTR, kb);
        #pragma unroll
        for (int mi = 0; mi < 4; ++mi) {
            v16bf afr = make_frag(Ab + (wm + mi * 16 + lr) * LDSTR, kb);
            #pragma unroll
            for (int ni = 0; ni < 2; ++ni)
                acc[mi][ni] = __builtin_amdgcn_wmma_f32_16x16x32_bf16(
                    false, afr, false, bfr[ni], (short)0, acc[mi][ni], false, false);
        }
        __syncthreads();   // reads of `cur` done before it is refilled next iter
    }
}

// ---------------- K0: fp32 -> bf16 convert ----------------
__global__ __launch_bounds__(256) void cvt_f32_bf16(const float* __restrict__ src,
                                                    bf16* __restrict__ dst, int n) {
    int i = blockIdx.x * 256 + threadIdx.x;
    if (i < n) dst[i] = (bf16)src[i];
}

// ---------------- K1: xz GEMM  C[M=NTOK, N=4096] = A[M,1024] @ W[N,1024]^T ----
__global__ __launch_bounds__(256) void gemm_xz(const bf16* __restrict__ A,
                                               const bf16* __restrict__ W,
                                               bf16* __restrict__ xs,
                                               bf16* __restrict__ zb) {
    __shared__ __align__(16) bf16 As[2 * BUFELE];
    __shared__ __align__(16) bf16 Bs[2 * BUFELE];

    const int m0   = blockIdx.y * 128;
    const int n0   = blockIdx.x * 128;
    const int lane = threadIdx.x & 31;
    const int wid  = threadIdx.x >> 5;
    const int wm   = (wid & 1) * 64;
    const int wn   = (wid >> 1) * 32;
    const int lr   = lane & 15;

    v8f acc[4][2];
    #pragma unroll
    for (int i = 0; i < 4; ++i)
        #pragma unroll
        for (int j = 0; j < 2; ++j) acc[i][j] = (v8f){};

    wmma_mainloop<D_MODEL>(A, W, m0, n0, As, Bs, acc);

    // Epilogue: split columns into xs (n < 2048) and z (n >= 2048), store bf16.
    #pragma unroll
    for (int mi = 0; mi < 4; ++mi)
        #pragma unroll
        for (int ni = 0; ni < 2; ++ni) {
            union { v8f v; float f[8]; } u; u.v = acc[mi][ni];
            const int mg = m0 + wm + mi * 16 + (lane >> 4) * 8;
            const int ng = n0 + wn + ni * 16 + lr;
            #pragma unroll
            for (int r = 0; r < 8; ++r) {
                bf16 val = (bf16)u.f[r];
                if (ng < D_INNER)
                    xs[(size_t)(mg + r) * D_INNER + ng] = val;
                else
                    zb[(size_t)(mg + r) * D_INNER + (ng - D_INNER)] = val;
            }
        }
}

// ---------------- K2: depthwise causal conv (width 4) + bias + SiLU ----------
__global__ __launch_bounds__(256) void conv_silu(const bf16* __restrict__ xs,
                                                 const float* __restrict__ cw,
                                                 const float* __restrict__ cb,
                                                 bf16* __restrict__ xc) {
    size_t idx = (size_t)blockIdx.x * 256 + threadIdx.x;  // over NTOK*D_INNER
    const int    c = (int)(idx % D_INNER);
    const size_t t = idx / D_INNER;       // global token (b*SEQ + l)
    const int    l = (int)(t % SEQ);
    float acc = cb[c];
    #pragma unroll
    for (int j = 0; j < 4; ++j) {
        int li = l - 3 + j;
        if (li >= 0)
            acc += cw[c * 4 + j] * (float)xs[(t - (size_t)(3 - j)) * D_INNER + c];
    }
    xc[idx] = (bf16)silu_f(acc);
}

// ---------------- K3: dt GEMM  dt[M,32] = xc[M,2048] @ Wxdt[32,2048]^T ------
__global__ __launch_bounds__(256) void gemm_dt(const bf16* __restrict__ xc,
                                               const bf16* __restrict__ Wxdt,
                                               bf16* __restrict__ dt) {
    const int tid  = threadIdx.x;
    const int lane = tid & 31;
    const int wid  = tid >> 5;
    const int m0   = blockIdx.x * 128 + wid * 16;
    const int kb   = (lane >> 4) * 8;
    const int lr   = lane & 15;

    v8f acc0 = (v8f){}, acc1 = (v8f){};
    for (int k0 = 0; k0 < D_INNER; k0 += KBLK) {
        v16bf a  = make_frag(&xc[(size_t)(m0 + lr) * D_INNER + k0], kb);
        v16bf b0 = make_frag(&Wxdt[(size_t)lr * D_INNER + k0], kb);
        v16bf b1 = make_frag(&Wxdt[(size_t)(16 + lr) * D_INNER + k0], kb);
        acc0 = __builtin_amdgcn_wmma_f32_16x16x32_bf16(false, a, false, b0, (short)0, acc0, false, false);
        acc1 = __builtin_amdgcn_wmma_f32_16x16x32_bf16(false, a, false, b1, (short)0, acc1, false, false);
    }
    union { v8f v; float f[8]; } u0, u1; u0.v = acc0; u1.v = acc1;
    const int mg = m0 + (lane >> 4) * 8;
    #pragma unroll
    for (int r = 0; r < 8; ++r) {
        dt[(size_t)(mg + r) * DT_RANK + lr]      = (bf16)u0.f[r];
        dt[(size_t)(mg + r) * DT_RANK + 16 + lr] = (bf16)u1.f[r];
    }
}

// ---------------- K4: a = softplus(dt @ Wdt^T + b_dt); y = xc * a * silu(z) --
__global__ __launch_bounds__(256) void a_gate_y(const bf16* __restrict__ dt,
                                                const bf16* __restrict__ Wdt,
                                                const float* __restrict__ bdt,
                                                const bf16* __restrict__ xc,
                                                const bf16* __restrict__ zb,
                                                bf16* __restrict__ y) {
    const int tid    = threadIdx.x;
    const int lane   = tid & 31;
    const int wid    = tid >> 5;
    const int m0     = blockIdx.x * 64 + (wid & 3) * 16;
    const int nstart = (wid >> 2) * 16;
    const int kb     = (lane >> 4) * 8;
    const int lr     = lane & 15;

    const v16bf afr = make_frag(&dt[(size_t)(m0 + lr) * DT_RANK], kb);  // K=32 total
    const int mg = m0 + (lane >> 4) * 8;

    for (int n0 = nstart; n0 < D_INNER; n0 += 32) {
        v16bf bfr = make_frag(&Wdt[(size_t)(n0 + lr) * DT_RANK], kb);
        v8f acc = (v8f){};
        acc = __builtin_amdgcn_wmma_f32_16x16x32_bf16(false, afr, false, bfr, (short)0, acc, false, false);
        union { v8f v; float f[8]; } u; u.v = acc;
        const int ng = n0 + lr;
        const float bias = bdt[ng];
        #pragma unroll
        for (int r = 0; r < 8; ++r) {
            size_t off = (size_t)(mg + r) * D_INNER + ng;
            float a  = softplus_f(u.f[r] + bias);
            float yy = (float)xc[off] * a * silu_f((float)zb[off]);
            y[off] = (bf16)yy;
        }
    }
}

// ---------------- K5: out GEMM  out[M,1024] = y[M,2048] @ Wout[1024,2048]^T + x
__global__ __launch_bounds__(256) void gemm_out(const bf16* __restrict__ Y,
                                                const bf16* __restrict__ W,
                                                const float* __restrict__ shortcut,
                                                float* __restrict__ out) {
    __shared__ __align__(16) bf16 As[2 * BUFELE];
    __shared__ __align__(16) bf16 Bs[2 * BUFELE];

    const int m0   = blockIdx.y * 128;
    const int n0   = blockIdx.x * 128;
    const int lane = threadIdx.x & 31;
    const int wid  = threadIdx.x >> 5;
    const int wm   = (wid & 1) * 64;
    const int wn   = (wid >> 1) * 32;
    const int lr   = lane & 15;

    v8f acc[4][2];
    #pragma unroll
    for (int i = 0; i < 4; ++i)
        #pragma unroll
        for (int j = 0; j < 2; ++j) acc[i][j] = (v8f){};

    wmma_mainloop<D_INNER>(Y, W, m0, n0, As, Bs, acc);

    #pragma unroll
    for (int mi = 0; mi < 4; ++mi)
        #pragma unroll
        for (int ni = 0; ni < 2; ++ni) {
            union { v8f v; float f[8]; } u; u.v = acc[mi][ni];
            const int mg = m0 + wm + mi * 16 + (lane >> 4) * 8;
            const int ng = n0 + wn + ni * 16 + lr;
            #pragma unroll
            for (int r = 0; r < 8; ++r) {
                size_t off = (size_t)(mg + r) * D_MODEL + ng;
                out[off] = u.f[r] + shortcut[off];
            }
        }
}

// ---------------- host-side launch ----------------
extern "C" void kernel_launch(void* const* d_in, const int* in_sizes, int n_in,
                              void* d_out, int out_size, void* d_ws, size_t ws_size,
                              hipStream_t stream) {
    const float* x      = (const float*)d_in[0];  // [B,L,1024]
    const float* W_in   = (const float*)d_in[1];  // [4096,1024]
    const float* conv_w = (const float*)d_in[2];  // [2048,1,4]
    const float* conv_b = (const float*)d_in[3];  // [2048]
    const float* W_x    = (const float*)d_in[4];  // [160,2048] (only rows 0..31 used)
    const float* W_dt   = (const float*)d_in[5];  // [2048,32]
    const float* b_dt   = (const float*)d_in[6];  // [2048]
    const float* W_out  = (const float*)d_in[7];  // [1024,2048]
    float* out = (float*)d_out;

    // Workspace layout (bf16 intermediates). ~250 MB total; y aliases xs.
    char* w = (char*)d_ws;
    bf16* x_bf    = (bf16*)w; w += (size_t)NTOK * D_MODEL * sizeof(bf16);   // 33.5 MB
    bf16* Win_bf  = (bf16*)w; w += (size_t)N_XZ * D_MODEL * sizeof(bf16);   //  8.4 MB
    bf16* Wxdt_bf = (bf16*)w; w += (size_t)DT_RANK * D_INNER * sizeof(bf16);
    bf16* Wdt_bf  = (bf16*)w; w += (size_t)D_INNER * DT_RANK * sizeof(bf16);
    bf16* Wout_bf = (bf16*)w; w += (size_t)D_MODEL * D_INNER * sizeof(bf16);//  4.2 MB
    bf16* xs_bf   = (bf16*)w; w += (size_t)NTOK * D_INNER * sizeof(bf16);   // 67 MB
    bf16* z_bf    = (bf16*)w; w += (size_t)NTOK * D_INNER * sizeof(bf16);   // 67 MB
    bf16* xc_bf   = (bf16*)w; w += (size_t)NTOK * D_INNER * sizeof(bf16);   // 67 MB
    bf16* dt_bf   = (bf16*)w; w += (size_t)NTOK * DT_RANK * sizeof(bf16);   //  1 MB
    bf16* y_bf    = xs_bf;  // xs dead after conv: reuse for gated y

    // K0: precision conversions (one-time streaming passes)
    {
        int n;
        n = NTOK * D_MODEL;    cvt_f32_bf16<<<(n + 255) / 256, 256, 0, stream>>>(x, x_bf, n);
        n = N_XZ * D_MODEL;    cvt_f32_bf16<<<(n + 255) / 256, 256, 0, stream>>>(W_in, Win_bf, n);
        n = DT_RANK * D_INNER; cvt_f32_bf16<<<(n + 255) / 256, 256, 0, stream>>>(W_x, Wxdt_bf, n);
        n = D_INNER * DT_RANK; cvt_f32_bf16<<<(n + 255) / 256, 256, 0, stream>>>(W_dt, Wdt_bf, n);
        n = D_MODEL * D_INNER; cvt_f32_bf16<<<(n + 255) / 256, 256, 0, stream>>>(W_out, Wout_bf, n);
    }

    // K1: xz GEMM -> xs, z
    gemm_xz<<<dim3(N_XZ / 128, NTOK / 128), 256, 0, stream>>>(x_bf, Win_bf, xs_bf, z_bf);

    // K2: depthwise causal conv + SiLU -> xc
    {
        size_t n = (size_t)NTOK * D_INNER;
        conv_silu<<<(unsigned)(n / 256), 256, 0, stream>>>(xs_bf, conv_w, conv_b, xc_bf);
    }

    // K3: dt GEMM
    gemm_dt<<<NTOK / 128, 256, 0, stream>>>(xc_bf, Wxdt_bf, dt_bf);

    // K4: a-GEMM + softplus + gating -> y (overwrites xs buffer)
    a_gate_y<<<NTOK / 64, 256, 0, stream>>>(dt_bf, Wdt_bf, b_dt, xc_bf, z_bf, y_bf);

    // K5: out GEMM + fp32 shortcut
    gemm_out<<<dim3(D_MODEL / 128, NTOK / 128), 256, 0, stream>>>(y_bf, Wout_bf, x, out);
}
